// MultiHeadAttention_48490180772138
// MI455X (gfx1250) — matrix operand order repaired and checked by reading, not compiled, not run on hardware
//
#include <hip/hip_runtime.h>
#include <hip/hip_bf16.h>

typedef __attribute__((ext_vector_type(16))) _Float16 v16h;
typedef __attribute__((ext_vector_type(8)))  _Float16 v8h;
typedef __attribute__((ext_vector_type(8)))  float    v8f;

#define NEGBIG (-1e9f)

__device__ __forceinline__ v8f wmma_f16(v16h a, v16h b, v8f c) {
    return __builtin_amdgcn_wmma_f32_16x16x32_f16(false, a, false, b, (short)0, c,
                                                  false, false);
}

// A fragment (16 rows x 32 K), row-major f16 source.
// ISA layout: lane m<16 -> row m, halves = K{0..7, 16..23};
//             lane m+16 -> row m, halves = K{8..15, 24..31}.
__device__ __forceinline__ v16h load_a_rm(const _Float16* __restrict__ base,
                                          int row_stride, int rowbase, int cbase) {
    const int lane = threadIdx.x & 31;
    const int m = lane & 15;
    const int hi = lane >> 4;
    const _Float16* p = base + (size_t)(rowbase + m) * row_stride + cbase;
    v8h lo = *(const v8h*)(p + (hi ? 8 : 0));
    v8h hh = *(const v8h*)(p + (hi ? 24 : 16));
    return __builtin_shufflevector(lo, hh, 0, 1, 2, 3, 4, 5, 6, 7,
                                   8, 9, 10, 11, 12, 13, 14, 15);
}

// Same A fragment but f32 source, converted to f16 on load.
__device__ __forceinline__ v16h load_a_f32(const float* __restrict__ base,
                                           int row_stride, int rowbase, int cbase) {
    const int lane = threadIdx.x & 31;
    const int m = lane & 15;
    const int hi = lane >> 4;
    const float* p = base + (size_t)(rowbase + m) * row_stride + cbase;
    const int o0 = hi ? 8 : 0, o1 = hi ? 24 : 16;
    v16h a;
#pragma unroll
    for (int i = 0; i < 8; ++i) {
        a[i]     = (_Float16)p[o0 + i];
        a[8 + i] = (_Float16)p[o1 + i];
    }
    return a;
}

// B fragment (32 K x 16 cols) from "col-major" storage src[col][c] (c contiguous).
// ISA layout: lane n<16 -> col n, halves = K{0..7, 16..23}; lane n+16 -> col n, K{8..15, 24..31}.
__device__ __forceinline__ v16h load_b_cm(const _Float16* __restrict__ base,
                                          int col_stride, int colbase, int cbase) {
    const int lane = threadIdx.x & 31;
    const int n = lane & 15;
    const int hi = lane >> 4;
    const _Float16* p = base + (size_t)(colbase + n) * col_stride + cbase + (hi ? 8 : 0);
    v8h lo = *(const v8h*)(p);
    v8h hh = *(const v8h*)(p + 16);
    return __builtin_shufflevector(lo, hh, 0, 1, 2, 3, 4, 5, 6, 7,
                                   8, 9, 10, 11, 12, 13, 14, 15);
}

__device__ __forceinline__ float rowmax16(float v) {
#pragma unroll
    for (int m = 1; m < 16; m <<= 1) v = fmaxf(v, __shfl_xor(v, m, 32));
    return v;
}
__device__ __forceinline__ float rowsum16(float v) {
#pragma unroll
    for (int m = 1; m < 16; m <<= 1) v += __shfl_xor(v, m, 32);
    return v;
}

// ---------------------------------------------------------------------------
// Kernel 1: weight prep.
//  WqT/WkT/WvT : f16 [64 n][1024 c]   (transposed, contraction-contiguous)
//  WoT         : f16 [1024 n][64 c], WoT[n][c] = sum_h Wo[h*64+c][n]
//  (heads are identical in the reference -> Wo block-row sum is exact)
// ---------------------------------------------------------------------------
__global__ void prep_weights(const float* __restrict__ Wq, const float* __restrict__ Wk,
                             const float* __restrict__ Wv, const float* __restrict__ Wo,
                             _Float16* __restrict__ WqT, _Float16* __restrict__ WkT,
                             _Float16* __restrict__ WvT, _Float16* __restrict__ WoT) {
    int idx = blockIdx.x * blockDim.x + threadIdx.x;
    if (idx >= 64 * 1024) return;
    int n = idx >> 10, c = idx & 1023;            // for 64x1024 transposes
    WqT[idx] = (_Float16)Wq[c * 64 + n];
    WkT[idx] = (_Float16)Wk[c * 64 + n];
    WvT[idx] = (_Float16)Wv[c * 64 + n];
    int nOut = idx >> 6, c2 = idx & 63;           // for 1024x64 folded Wo
    float s = 0.f;
#pragma unroll
    for (int h = 0; h < 16; ++h) s += Wo[(h * 64 + c2) * 1024 + nOut];
    WoT[idx] = (_Float16)s;
}

// ---------------------------------------------------------------------------
// Kernel 2: QKV projection. grid (64 row-tiles, 3 mats), 256 thr (8 waves).
// Wave computes a 16x64 output tile; K loop over 1024 unrolled by 64 with two
// ping-pong fragment sets: each set is consumed by 4 WMMAs and then reloaded
// in place for the iteration after next, so ~10 b128 loads stay in flight
// with no register-rotation copies. V output is written transposed.
// ---------------------------------------------------------------------------
__global__ __launch_bounds__(256) void qkv_proj(
    const float* __restrict__ x, const float* __restrict__ y,
    const float* __restrict__ bq, const float* __restrict__ bk,
    const float* __restrict__ bv,
    const _Float16* __restrict__ WqT, const _Float16* __restrict__ WkT,
    const _Float16* __restrict__ WvT,
    _Float16* __restrict__ qf, _Float16* __restrict__ kf,
    _Float16* __restrict__ vT) {
    const int z = blockIdx.y;
    const int wave = threadIdx.x >> 5;
    const int lane = threadIdx.x & 31;
    const int rowbase = blockIdx.x * 128 + wave * 16;   // [0, 8192)
    const float* src = (z == 0) ? x : y;
    const _Float16* WT = (z == 0) ? WqT : (z == 1) ? WkT : WvT;
    const float* bias = (z == 0) ? bq : (z == 1) ? bk : bv;

    // per-lane row pointer for prefetching the streamed activations
    const float* prefp = src + (size_t)(rowbase + (lane & 15)) * 1024;

    v8f acc[4];
#pragma unroll
    for (int nt = 0; nt < 4; ++nt) acc[nt] = (v8f)(0.f);

    // ---- prologue: two fragment sets (kc = 0 and kc = 32) ----
    v16h aA  = load_a_f32(src, 1024, rowbase, 0);
    v16h bA0 = load_b_cm(WT, 1024, 0,  0);
    v16h bA1 = load_b_cm(WT, 1024, 16, 0);
    v16h bA2 = load_b_cm(WT, 1024, 32, 0);
    v16h bA3 = load_b_cm(WT, 1024, 48, 0);
    v16h aB  = load_a_f32(src, 1024, rowbase, 32);
    v16h bB0 = load_b_cm(WT, 1024, 0,  32);
    v16h bB1 = load_b_cm(WT, 1024, 16, 32);
    v16h bB2 = load_b_cm(WT, 1024, 32, 32);
    v16h bB3 = load_b_cm(WT, 1024, 48, 32);

    // steady state: kc = 0..896; consume a set, then reload it for kc+64/kc+96
    for (int kc = 0; kc + 64 < 1024; kc += 64) {
        acc[0] = wmma_f16(aA, bA0, acc[0]);
        acc[1] = wmma_f16(aA, bA1, acc[1]);
        acc[2] = wmma_f16(aA, bA2, acc[2]);
        acc[3] = wmma_f16(aA, bA3, acc[3]);
        __builtin_prefetch(prefp + kc + 128, 0, 3);      // global_prefetch_b8
        aA  = load_a_f32(src, 1024, rowbase, kc + 64);
        bA0 = load_b_cm(WT, 1024, 0,  kc + 64);
        bA1 = load_b_cm(WT, 1024, 16, kc + 64);
        bA2 = load_b_cm(WT, 1024, 32, kc + 64);
        bA3 = load_b_cm(WT, 1024, 48, kc + 64);

        acc[0] = wmma_f16(aB, bB0, acc[0]);
        acc[1] = wmma_f16(aB, bB1, acc[1]);
        acc[2] = wmma_f16(aB, bB2, acc[2]);
        acc[3] = wmma_f16(aB, bB3, acc[3]);
        aB  = load_a_f32(src, 1024, rowbase, kc + 96);
        bB0 = load_b_cm(WT, 1024, 0,  kc + 96);
        bB1 = load_b_cm(WT, 1024, 16, kc + 96);
        bB2 = load_b_cm(WT, 1024, 32, kc + 96);
        bB3 = load_b_cm(WT, 1024, 48, kc + 96);
    }
    // epilogue: kc = 960 (set A) and kc = 992 (set B)
    acc[0] = wmma_f16(aA, bA0, acc[0]);
    acc[1] = wmma_f16(aA, bA1, acc[1]);
    acc[2] = wmma_f16(aA, bA2, acc[2]);
    acc[3] = wmma_f16(aA, bA3, acc[3]);
    acc[0] = wmma_f16(aB, bB0, acc[0]);
    acc[1] = wmma_f16(aB, bB1, acc[1]);
    acc[2] = wmma_f16(aB, bB2, acc[2]);
    acc[3] = wmma_f16(aB, bB3, acc[3]);

    const int nlo = lane & 15;
    const int hb = (lane >> 4) * 8;   // C layout: vgpr r = row r (+8 for hi lanes)
    if (z == 0) {
#pragma unroll
        for (int nt = 0; nt < 4; ++nt) {
            int col = nt * 16 + nlo;
            float bc = bias[col];
#pragma unroll
            for (int r = 0; r < 8; ++r)
                qf[(size_t)(rowbase + r + hb) * 64 + col] =
                    (_Float16)(acc[nt][r] + bc);
        }
    } else if (z == 1) {
#pragma unroll
        for (int nt = 0; nt < 4; ++nt) {
            int col = nt * 16 + nlo;
            float bc = bias[col];
#pragma unroll
            for (int r = 0; r < 8; ++r)
                kf[(size_t)(rowbase + r + hb) * 64 + col] =
                    (_Float16)(acc[nt][r] + bc);
        }
    } else {
#pragma unroll
        for (int nt = 0; nt < 4; ++nt) {
            int col = nt * 16 + nlo;
            float bc = bias[col];
#pragma unroll
            for (int r = 0; r < 8; ++r) {
                int rg = rowbase + r + hb;
                int bI = rg >> 11, s = rg & 2047;
                vT[(((size_t)bI * 64 + col) << 11) + s] =
                    (_Float16)(acc[nt][r] + bc);
            }
        }
    }
}

// ---------------------------------------------------------------------------
// Kernel 3: causal flash attention. grid 64 (= 4 batch x 16 q-blocks of 128),
// 8 waves; each wave owns 16 query rows, streams keys in blocks of 32 with
// online softmax. V-fragment loads are issued before the softmax VALU block
// so their latency is covered. P goes C-layout -> A-layout through per-wave
// LDS scratch (per-wave region; LDS ops are in-order within a wave).
// ---------------------------------------------------------------------------
__global__ __launch_bounds__(256) void attn(
    const _Float16* __restrict__ qf, const _Float16* __restrict__ kf,
    const _Float16* __restrict__ vT, _Float16* __restrict__ ho) {
    __shared__ __align__(16) _Float16 ldsP[8 * 16 * 32];

    const int wave = threadIdx.x >> 5;
    const int lane = threadIdx.x & 31;
    const int b = blockIdx.x >> 4;
    const int qblk = blockIdx.x & 15;
    const int qb = qblk * 128 + wave * 16;       // seq position of q tile
    const int grow = b * 2048 + qb;              // global row base
    const int nlo = lane & 15;
    const int hb = (lane >> 4) * 8;
    const float scale = 0.125f;                  // 1/sqrt(64)

    const _Float16* kbase = kf + (size_t)b * 2048 * 64;  // [s][64]
    const _Float16* vbase = vT + (size_t)b * 64 * 2048;  // [d][s]
    _Float16* myP = ldsP + wave * 512;                   // 16 x 32, stride 32

    v16h qa0 = load_a_rm(qf, 64, grow, 0);
    v16h qa1 = load_a_rm(qf, 64, grow, 32);

    v8f o[4];
#pragma unroll
    for (int nt = 0; nt < 4; ++nt) o[nt] = (v8f)(0.f);
    float mrow[8], lrow[8];
#pragma unroll
    for (int r = 0; r < 8; ++r) { mrow[r] = -1e30f; lrow[r] = 0.f; }

    for (int jb = 0; jb < qb + 16; jb += 32) {
        // ---- S = Q K^T for keys [jb, jb+32): batch the 4 K-frag loads ----
        v16h kb00 = load_b_cm(kbase, 64, jb,      0);
        v16h kb01 = load_b_cm(kbase, 64, jb,      32);
        v16h kb10 = load_b_cm(kbase, 64, jb + 16, 0);
        v16h kb11 = load_b_cm(kbase, 64, jb + 16, 32);
        v8f s0 = (v8f)(0.f), s1 = (v8f)(0.f);
        s0 = wmma_f16(qa0, kb00, s0);
        s0 = wmma_f16(qa1, kb01, s0);
        s1 = wmma_f16(qa0, kb10, s1);
        s1 = wmma_f16(qa1, kb11, s1);

        // ---- issue V-frag loads early; consumed after the softmax VALU ----
        v16h vb0 = load_b_cm(vbase, 2048, 0,  jb);
        v16h vb1 = load_b_cm(vbase, 2048, 16, jb);
        v16h vb2 = load_b_cm(vbase, 2048, 32, jb);
        v16h vb3 = load_b_cm(vbase, 2048, 48, jb);

        // ---- online softmax, C layout: row r (+8 for hi lanes), col = lane&15
#pragma unroll
        for (int r = 0; r < 8; ++r) {
            int qrow = qb + r + hb;
            float a0 = s0[r] * scale;
            float a1 = s1[r] * scale;
            if (jb + nlo > qrow)      a0 = NEGBIG;   // causal mask
            if (jb + 16 + nlo > qrow) a1 = NEGBIG;
            float mt = rowmax16(fmaxf(a0, a1));
            float mold = mrow[r];
            float mnew = fmaxf(mold, mt);
            float f = __expf(mold - mnew);
            float p0 = __expf(a0 - mnew);
            float p1 = __expf(a1 - mnew);
            float ls = rowsum16(p0 + p1);
            mrow[r] = mnew;
            lrow[r] = lrow[r] * f + ls;
#pragma unroll
            for (int nt = 0; nt < 4; ++nt) o[nt][r] *= f;
            int prow = r + hb;
            myP[prow * 32 + nlo]      = (_Float16)p0;
            myP[prow * 32 + 16 + nlo] = (_Float16)p1;
        }

        // ---- O += P V ----
        v16h pa = load_a_rm(myP, 32, 0, 0);
        o[0] = wmma_f16(pa, vb0, o[0]);
        o[1] = wmma_f16(pa, vb1, o[1]);
        o[2] = wmma_f16(pa, vb2, o[2]);
        o[3] = wmma_f16(pa, vb3, o[3]);
    }

#pragma unroll
    for (int nt = 0; nt < 4; ++nt)
#pragma unroll
        for (int r = 0; r < 8; ++r) {
            float v = o[nt][r] / lrow[r];
            ho[(size_t)(grow + r + hb) * 64 + nt * 16 + nlo] = (_Float16)v;
        }
}

// ---------------------------------------------------------------------------
// Kernel 4: out = head_out @ Wo_sum + bo.  grid (64 row-tiles, 16 col-tiles).
// ---------------------------------------------------------------------------
__global__ __launch_bounds__(256) void out_proj(
    const _Float16* __restrict__ ho, const _Float16* __restrict__ WoT,
    const float* __restrict__ bo, float* __restrict__ out) {
    const int wave = threadIdx.x >> 5;
    const int lane = threadIdx.x & 31;
    const int rowbase = blockIdx.x * 128 + wave * 16;
    const int colbase = blockIdx.y * 64;

    v16h a0 = load_a_rm(ho, 64, rowbase, 0);
    v16h a1 = load_a_rm(ho, 64, rowbase, 32);
    v16h b00 = load_b_cm(WoT, 64, colbase,      0);
    v16h b01 = load_b_cm(WoT, 64, colbase,      32);
    v16h b10 = load_b_cm(WoT, 64, colbase + 16, 0);
    v16h b11 = load_b_cm(WoT, 64, colbase + 16, 32);
    v16h b20 = load_b_cm(WoT, 64, colbase + 32, 0);
    v16h b21 = load_b_cm(WoT, 64, colbase + 32, 32);
    v16h b30 = load_b_cm(WoT, 64, colbase + 48, 0);
    v16h b31 = load_b_cm(WoT, 64, colbase + 48, 32);

    v8f acc[4];
#pragma unroll
    for (int nt = 0; nt < 4; ++nt) acc[nt] = (v8f)(0.f);
    acc[0] = wmma_f16(a0, b00, acc[0]);
    acc[0] = wmma_f16(a1, b01, acc[0]);
    acc[1] = wmma_f16(a0, b10, acc[1]);
    acc[1] = wmma_f16(a1, b11, acc[1]);
    acc[2] = wmma_f16(a0, b20, acc[2]);
    acc[2] = wmma_f16(a1, b21, acc[2]);
    acc[3] = wmma_f16(a0, b30, acc[3]);
    acc[3] = wmma_f16(a1, b31, acc[3]);

    const int nlo = lane & 15;
    const int hb = (lane >> 4) * 8;
#pragma unroll
    for (int nt = 0; nt < 4; ++nt) {
        int col = colbase + nt * 16 + nlo;
        float bc = bo[col];
#pragma unroll
        for (int r = 0; r < 8; ++r)
            out[(size_t)(rowbase + r + hb) * 1024 + col] = acc[nt][r] + bc;
    }
}

// ---------------------------------------------------------------------------
extern "C" void kernel_launch(void* const* d_in, const int* in_sizes, int n_in,
                              void* d_out, int out_size, void* d_ws, size_t ws_size,
                              hipStream_t stream) {
    const float* x  = (const float*)d_in[0];
    const float* y  = (const float*)d_in[1];
    // d_in[2] = max_seq_length (scalar), unused: S hardcoded to 2048
    const float* Wq = (const float*)d_in[3];
    const float* bq = (const float*)d_in[4];
    const float* Wk = (const float*)d_in[5];
    const float* bk = (const float*)d_in[6];
    const float* Wv = (const float*)d_in[7];
    const float* bv = (const float*)d_in[8];
    const float* Wo = (const float*)d_in[9];
    const float* bo = (const float*)d_in[10];
    float* out = (float*)d_out;

    char* ws = (char*)d_ws;
    const size_t MB = 1u << 20;
    _Float16* qf  = (_Float16*)(ws + 0 * MB);            // [8192][64]
    _Float16* kf  = (_Float16*)(ws + 1 * MB);            // [8192][64]
    _Float16* vT  = (_Float16*)(ws + 2 * MB);            // [4][64][2048]
    _Float16* ho  = (_Float16*)(ws + 3 * MB);            // [8192][64]
    _Float16* WqT = (_Float16*)(ws + 4 * MB);            // [64][1024]
    _Float16* WkT = (_Float16*)(ws + 4 * MB + 128 * 1024);
    _Float16* WvT = (_Float16*)(ws + 4 * MB + 256 * 1024);
    _Float16* WoT = (_Float16*)(ws + 4 * MB + 384 * 1024); // [1024][64]

    prep_weights<<<256, 256, 0, stream>>>(Wq, Wk, Wv, Wo, WqT, WkT, WvT, WoT);
    qkv_proj<<<dim3(64, 3), 256, 0, stream>>>(x, y, bq, bk, bv, WqT, WkT, WvT,
                                              qf, kf, vT);
    attn<<<64, 256, 0, stream>>>(qf, kf, vT, ho);
    out_proj<<<dim3(64, 16), 256, 0, stream>>>(ho, WoT, bo, out);
}
